// Decoder_69801808495349
// MI455X (gfx1250) — matrix operand order repaired and checked by reading, not compiled, not run on hardware
//
#include <hip/hip_runtime.h>
#include <hip/hip_bf16.h>
#include <stdint.h>

// ---------------- problem constants (match reference) ----------------
#define L_LAYERS 4
#define D_    1024
#define H_    16
#define HD_   64
#define FFN_  4096
#define B_    4
#define LT_   1024
#define LS_   1024
#define M_    (B_ * LT_)      // 4096 rows in every GEMM (B*L)
#define NEG_  (-1e9f)
#define EPS_  (1e-6f)

// ---------------- CDNA5 WMMA / TDM types ----------------
typedef __bf16        bf16x16 __attribute__((ext_vector_type(16)));
typedef float         f32x8   __attribute__((ext_vector_type(8)));
typedef unsigned int  u32x4   __attribute__((ext_vector_type(4)));
typedef int           i32x8   __attribute__((ext_vector_type(8)));
typedef int           i32x4   __attribute__((ext_vector_type(4)));

union Frag16 { bf16x16 v; uint32_t u[8]; };

__device__ __forceinline__ uint16_t f2bf(float f) {
    __bf16 h = (__bf16)f;                    // RNE hardware convert
    return __builtin_bit_cast(uint16_t, h);
}

// ---------------- embed gather ----------------
__global__ void k_embed(const int* __restrict__ ids, const float* __restrict__ emb,
                        float* __restrict__ x) {
    int row = blockIdx.x;                 // 0..B*LT-1
    int id  = ids[row];
    const float4* src = (const float4*)(emb + (size_t)id * D_);
    float4*       dst = (float4*)(x + (size_t)row * D_);
    dst[threadIdx.x] = src[threadIdx.x];  // 256 thr * 4 = 1024
}

// ---------------- fp32 -> bf16 elementwise ----------------
__global__ void k_cvt_bf16(const float* __restrict__ in, uint16_t* __restrict__ out, int n) {
    int i = blockIdx.x * blockDim.x + threadIdx.x;
    if (i < n) out[i] = f2bf(in[i]);
}

// ---------------- RMSNorm (one block per row, D=1024) ----------------
template<bool BF16OUT>
__global__ void k_rms(const float* __restrict__ x, const float* __restrict__ g,
                      void* __restrict__ out) {
    __shared__ float red[256];
    int row = blockIdx.x, t = threadIdx.x;
    const float4* xr = (const float4*)(x + (size_t)row * D_);
    float4 v = xr[t];
    red[t] = v.x*v.x + v.y*v.y + v.z*v.z + v.w*v.w;
    __syncthreads();
    for (int s = 128; s > 0; s >>= 1) {
        if (t < s) red[t] += red[t + s];
        __syncthreads();
    }
    float inv = rsqrtf(red[0] * (1.0f / D_) + EPS_);
    float4 gv = ((const float4*)g)[t];
    float o0 = v.x * inv * gv.x, o1 = v.y * inv * gv.y;
    float o2 = v.z * inv * gv.z, o3 = v.w * inv * gv.w;
    if (BF16OUT) {
        uint16_t* op = (uint16_t*)out + (size_t)row * D_ + t * 4;
        op[0] = f2bf(o0); op[1] = f2bf(o1); op[2] = f2bf(o2); op[3] = f2bf(o3);
    } else {
        float4 o; o.x = o0; o.y = o1; o.z = o2; o.w = o3;
        ((float4*)out)[(size_t)row * 256 + t] = o;
    }
}

// ---------------- bf16 WMMA GEMM: C[M,N] = A_bf16[M,K] @ W_f32[K,N] ----------------
// A tile staged via Tensor Data Mover (async DMA, TENSORcnt); W tile converted
// fp32->bf16 in-flight and stored transposed [n][k] for B-fragment reads.
// MODE 0: outF = acc + res (fp32, in-place residual)
// MODE 1: outH = bf16(acc) row-major, optional ReLU
// MODE 2: outH = bf16(acc) at [B,H,L,HD]   (Q / K layout)
// MODE 3: outH = bf16(acc) at [B,H,HD,L]   (V layout: K-pairs of P@V contiguous)
template<int MODE>
__launch_bounds__(128)
__global__ void k_gemm(const uint16_t* __restrict__ A, const float* __restrict__ W,
                       int N, int K, const float* __restrict__ res,
                       float* __restrict__ outF, uint16_t* __restrict__ outH, int relu) {
    __shared__ uint16_t ASH[128 * 32];   // A tile  [row][k]   (TDM destination)
    __shared__ uint16_t WSH[128 * 32];   // W tile  [n][k]     (transposed for B frags)

    const int t    = threadIdx.x;
    const int lane = t & 31;
    const int wave = t >> 5;
    const int wr   = wave >> 1, wc = wave & 1;   // 2x2 waves -> 128x128 block
    const int m16  = lane & 15, kh = lane >> 4;
    const int rowBase = blockIdx.y * 128;
    const int colBase = blockIdx.x * 128;
    const uint32_t ldsOff = (uint32_t)(uintptr_t)&ASH[0];   // low 32b = LDS byte offset

    f32x8 acc[4][4];
    const f32x8 z8 = {0.f,0.f,0.f,0.f,0.f,0.f,0.f,0.f};
    for (int i = 0; i < 4; ++i)
        for (int j = 0; j < 4; ++j) acc[i][j] = z8;

    for (int k0 = 0; k0 < K; k0 += 32) {
        __syncthreads();
        // ---- A tile: 128 rows x 32 k bf16 via TDM (wave 0 issues; EXEC ignored) ----
        if (t < 32) {
            uint64_t gaddr = (uint64_t)(uintptr_t)(A + (size_t)rowBase * K + k0);
            u32x4 g0;
            g0[0] = 1u;                                           // count=1, user D#
            g0[1] = ldsOff;                                       // lds_addr
            g0[2] = (uint32_t)gaddr;                              // global_addr lo
            g0[3] = (uint32_t)((gaddr >> 32) & 0x1FFFFFFu) | (2u << 30); // addr hi | type=2
            i32x8 g1;
            g1[0] = (int)(1u << 16);                              // data_size=2B
            g1[1] = (int)(((uint32_t)K & 0xFFFFu) << 16);         // tensor_dim0[15:0]
            g1[2] = (int)(((uint32_t)K >> 16) | (((uint32_t)M_ & 0xFFFFu) << 16)); // dim0 hi | dim1 lo
            g1[3] = (int)(32u << 16);                             // dim1 hi=0 | tile_dim0=32
            g1[4] = 128;                                          // tile_dim1=128, tile_dim2=0
            g1[5] = K;                                            // tensor_dim0_stride lo
            g1[6] = 0;
            g1[7] = 0;
            i32x4 zz; zz[0] = 0; zz[1] = 0; zz[2] = 0; zz[3] = 0;
            i32x8 zz8; zz8[0]=0; zz8[1]=0; zz8[2]=0; zz8[3]=0; zz8[4]=0; zz8[5]=0; zz8[6]=0; zz8[7]=0;
            __builtin_amdgcn_tensor_load_to_lds(g0, g1, zz, zz, zz8, 0);
        }
        // ---- prefetch next W K-slice into GL2 (weights are re-read by 32 row-blocks) ----
        if (k0 + 32 < K)
            __builtin_prefetch(&W[(size_t)(k0 + 32) * N + colBase], 0, 1);
        // ---- W tile: 32 k x 128 n fp32 -> bf16, transposed to [n][k] ----
        #pragma unroll
        for (int i = 0; i < 8; ++i) {
            int idx = i * 128 + t;
            int kk = idx >> 5;
            int nn = (idx & 31) << 2;
            const float4 w4 = *(const float4*)&W[(size_t)(k0 + kk) * N + colBase + nn];
            WSH[(nn + 0) * 32 + kk] = f2bf(w4.x);
            WSH[(nn + 1) * 32 + kk] = f2bf(w4.y);
            WSH[(nn + 2) * 32 + kk] = f2bf(w4.z);
            WSH[(nn + 3) * 32 + kk] = f2bf(w4.w);
        }
        if (t < 32) __builtin_amdgcn_s_wait_tensorcnt(0);
        __syncthreads();

        Frag16 a[4], b[4];
        #pragma unroll
        for (int mi = 0; mi < 4; ++mi)
            #pragma unroll
            for (int j = 0; j < 8; ++j) {
                // 16-bit A 16x32 layout: VGPR j<4 -> K=2j..2j+1 (+8*lanehalf); j>=4 -> +16
                int kb = (j < 4 ? 2 * j : 16 + 2 * (j - 4)) + 8 * kh;
                a[mi].u[j] = *(const uint32_t*)&ASH[(wr * 64 + mi * 16 + m16) * 32 + kb];
            }
        #pragma unroll
        for (int ni = 0; ni < 4; ++ni)
            #pragma unroll
            for (int j = 0; j < 8; ++j) {
                // B 32x16: lane = column, VGPR j<4 -> K=2j..2j+1 (+8*lanehalf); j>=4 -> +16
                int kb = (j < 4 ? 2 * j : 16 + 2 * (j - 4)) + 8 * kh;
                b[ni].u[j] = *(const uint32_t*)&WSH[(wc * 64 + ni * 16 + m16) * 32 + kb];
            }
        #pragma unroll
        for (int mi = 0; mi < 4; ++mi)
            #pragma unroll
            for (int ni = 0; ni < 4; ++ni)
                acc[mi][ni] = __builtin_amdgcn_wmma_f32_16x16x32_bf16(
                    false, a[mi].v, false, b[ni].v, (short)0, acc[mi][ni], false, false);
    }

    // epilogue: C layout M = r + 8*(lane>=16), N = lane&15
    #pragma unroll
    for (int mi = 0; mi < 4; ++mi)
        #pragma unroll
        for (int ni = 0; ni < 4; ++ni)
            #pragma unroll
            for (int r = 0; r < 8; ++r) {
                int Mg = rowBase + wr * 64 + mi * 16 + r + 8 * kh;
                int Ng = colBase + wc * 64 + ni * 16 + m16;
                float val = acc[mi][ni][r];
                if (MODE == 0) {
                    size_t o = (size_t)Mg * N + Ng;
                    outF[o] = val + res[o];
                } else if (MODE == 1) {
                    if (relu) val = fmaxf(val, 0.f);
                    outH[(size_t)Mg * N + Ng] = f2bf(val);
                } else if (MODE == 2) {
                    int bb = Mg >> 10, ll = Mg & 1023;
                    int hh = Ng >> 6,  hd = Ng & 63;
                    outH[(((size_t)(bb * H_ + hh)) * LT_ + ll) * HD_ + hd] = f2bf(val);
                } else {
                    int bb = Mg >> 10, ll = Mg & 1023;
                    int hh = Ng >> 6,  hd = Ng & 63;
                    outH[(((size_t)(bb * H_ + hh)) * HD_ + hd) * LT_ + ll] = f2bf(val);
                }
            }
}

// ---------------- flash attention: one wave per (b, h, 16-query tile) ----------------
// Q,K: [B,H,L,64] bf16; V: [B,H,64,L] bf16; O: [B,L,D] bf16
__launch_bounds__(32)
__global__ void k_attn(const uint16_t* __restrict__ Q, const uint16_t* __restrict__ Kt,
                       const uint16_t* __restrict__ Vt, uint16_t* __restrict__ O,
                       int causal) {
    __shared__ float S[16][32];
    __shared__ float rowM[16];
    __shared__ float rowSc[16];

    const int lane = threadIdx.x;
    const int m16 = lane & 15, kh = lane >> 4;
    const int b = blockIdx.z, h = blockIdx.y;
    const int q0 = blockIdx.x * 16;
    const uint16_t* q = Q  + (size_t)(b * H_ + h) * LT_ * HD_;
    const uint16_t* k = Kt + (size_t)(b * H_ + h) * LT_ * HD_;
    const uint16_t* v = Vt + (size_t)(b * H_ + h) * HD_ * LT_;

    // Q A-fragments for head-dim halves (K-step 32, HD=64 -> 2 frags)
    Frag16 aq[2];
    #pragma unroll
    for (int s = 0; s < 2; ++s)
        #pragma unroll
        for (int j = 0; j < 8; ++j) {
            int kb = (j < 4 ? 2 * j : 16 + 2 * (j - 4)) + 8 * kh + s * 32;
            aq[s].u[j] = *(const uint32_t*)&q[(size_t)(q0 + m16) * HD_ + kb];
        }

    f32x8 oacc[4];
    const f32x8 z8 = {0.f,0.f,0.f,0.f,0.f,0.f,0.f,0.f};
    for (int i = 0; i < 4; ++i) oacc[i] = z8;
    float mcur = -3.0e38f, lcur = 0.f;

    const int kend = causal ? (q0 + 16) : LT_;
    for (int kb0 = 0; kb0 < kend; kb0 += 32) {
        // ---- scores S[16 x 32] = (Q Kt)/8 (+ mask) ----
        #pragma unroll
        for (int sub = 0; sub < 2; ++sub) {
            f32x8 c = z8;
            #pragma unroll
            for (int s = 0; s < 2; ++s) {
                Frag16 bk;
                #pragma unroll
                for (int j = 0; j < 8; ++j) {
                    int kk = (j < 4 ? 2 * j : 16 + 2 * (j - 4)) + 8 * kh + s * 32;   // K = head-dim
                    bk.u[j] = *(const uint32_t*)&k[(size_t)(kb0 + sub * 16 + m16) * HD_ + kk];
                }
                c = __builtin_amdgcn_wmma_f32_16x16x32_bf16(
                        false, aq[s].v, false, bk.v, (short)0, c, false, false);
            }
            #pragma unroll
            for (int r = 0; r < 8; ++r) {
                int Mq = q0 + r + 8 * kh;
                int Nk = kb0 + sub * 16 + m16;
                float sc = c[r] * 0.125f;               // 1/sqrt(64)
                if (causal && Nk > Mq) sc = NEG_;
                S[r + 8 * kh][sub * 16 + m16] = sc;
            }
        }
        __syncthreads();
        // ---- online softmax stats: lanes 0..15 own rows ----
        if (lane < 16) {
            float mb = -3.0e38f;
            #pragma unroll
            for (int kk = 0; kk < 32; ++kk) mb = fmaxf(mb, S[lane][kk]);
            float mnew = fmaxf(mcur, mb);
            float scl  = __expf(mcur - mnew);
            float ssum = 0.f;
            #pragma unroll
            for (int kk = 0; kk < 32; ++kk) ssum += __expf(S[lane][kk] - mnew);
            lcur = lcur * scl + ssum;
            mcur = mnew;
            rowM[lane] = mnew; rowSc[lane] = scl;
        }
        __syncthreads();
        // ---- P (bf16 A-frag 16x32) ----
        Frag16 p;
        #pragma unroll
        for (int j = 0; j < 8; ++j) {
            int ke = (j < 4 ? 2 * j : 16 + 2 * (j - 4)) + 8 * kh;
            float e0 = __expf(S[m16][ke]     - rowM[m16]);
            float e1 = __expf(S[m16][ke + 1] - rowM[m16]);
            p.u[j] = (uint32_t)f2bf(e0) | ((uint32_t)f2bf(e1) << 16);
        }
        // ---- rescale O and accumulate P @ V ----
        #pragma unroll
        for (int ni = 0; ni < 4; ++ni) {
            #pragma unroll
            for (int r = 0; r < 8; ++r) oacc[ni][r] *= rowSc[r + 8 * kh];
            Frag16 bv;
            #pragma unroll
            for (int j = 0; j < 8; ++j) {
                int kk = (j < 4 ? 2 * j : 16 + 2 * (j - 4)) + 8 * kh;               // K = key index in block
                bv.u[j] = *(const uint32_t*)&v[(size_t)(ni * 16 + m16) * LT_ + kb0 + kk];
            }
            oacc[ni] = __builtin_amdgcn_wmma_f32_16x16x32_bf16(
                           false, p.v, false, bv.v, (short)0, oacc[ni], false, false);
        }
        __syncthreads();
    }
    if (lane < 16) rowSc[lane] = 1.f / lcur;
    __syncthreads();
    #pragma unroll
    for (int ni = 0; ni < 4; ++ni)
        #pragma unroll
        for (int r = 0; r < 8; ++r) {
            float o = oacc[ni][r] * rowSc[r + 8 * kh];
            int row = q0 + r + 8 * kh;
            int col = h * HD_ + ni * 16 + m16;
            O[((size_t)b * LT_ + row) * D_ + col] = f2bf(o);
        }
}

// ---------------- orchestration ----------------
extern "C" void kernel_launch(void* const* d_in, const int* in_sizes, int n_in,
                              void* d_out, int out_size, void* d_ws, size_t ws_size,
                              hipStream_t stream) {
    (void)in_sizes; (void)n_in; (void)out_size; (void)ws_size;
    const int*   ids    = (const int*)  d_in[0];
    const float* memory = (const float*)d_in[1];
    const float* embed  = (const float*)d_in[2];
    const float* sa_g   = (const float*)d_in[3];
    const float* sa_wq  = (const float*)d_in[4];
    const float* sa_wk  = (const float*)d_in[5];
    const float* sa_wv  = (const float*)d_in[6];
    const float* sa_wo  = (const float*)d_in[7];
    const float* ca_g   = (const float*)d_in[8];
    const float* ca_wq  = (const float*)d_in[9];
    const float* ca_wk  = (const float*)d_in[10];
    const float* ca_wv  = (const float*)d_in[11];
    const float* ca_wo  = (const float*)d_in[12];
    const float* mlp_g  = (const float*)d_in[13];
    const float* w1     = (const float*)d_in[14];
    const float* w2     = (const float*)d_in[15];
    const float* fin_g  = (const float*)d_in[16];
    float* out = (float*)d_out;

    char* ws = (char*)d_ws;
    size_t off = 0;
    float*    x    = (float*)   (ws + off); off += (size_t)M_ * D_   * 4;
    uint16_t* hbf  = (uint16_t*)(ws + off); off += (size_t)M_ * D_   * 2;
    uint16_t* qb   = (uint16_t*)(ws + off); off += (size_t)M_ * D_   * 2;
    uint16_t* kbuf = (uint16_t*)(ws + off); off += (size_t)M_ * D_   * 2;
    uint16_t* vbuf = (uint16_t*)(ws + off); off += (size_t)M_ * D_   * 2;
    uint16_t* att  = (uint16_t*)(ws + off); off += (size_t)M_ * D_   * 2;
    uint16_t* memb = (uint16_t*)(ws + off); off += (size_t)M_ * D_   * 2;
    uint16_t* ffn  = (uint16_t*)(ws + off); off += (size_t)M_ * FFN_ * 2;

    dim3 g1024(D_ / 128,   M_ / 128);   // (8, 32)
    dim3 g4096(FFN_ / 128, M_ / 128);   // (32, 32)
    dim3 gattn(LT_ / 16, H_, B_);

    k_embed<<<M_, 256, 0, stream>>>(ids, embed, x);
    k_cvt_bf16<<<(M_ * D_) / 256, 256, 0, stream>>>(memory, memb, M_ * D_);

    for (int l = 0; l < L_LAYERS; ++l) {
        const size_t wofs = (size_t)l * D_ * D_;
        // ---- self attention ----
        k_rms<true><<<M_, 256, 0, stream>>>(x, sa_g + l * D_, (void*)hbf);
        k_gemm<2><<<g1024, 128, 0, stream>>>(hbf, sa_wq + wofs, D_, D_, nullptr, nullptr, qb,   0);
        k_gemm<2><<<g1024, 128, 0, stream>>>(hbf, sa_wk + wofs, D_, D_, nullptr, nullptr, kbuf, 0);
        k_gemm<3><<<g1024, 128, 0, stream>>>(hbf, sa_wv + wofs, D_, D_, nullptr, nullptr, vbuf, 0);
        k_attn<<<gattn, 32, 0, stream>>>(qb, kbuf, vbuf, att, 1);
        k_gemm<0><<<g1024, 128, 0, stream>>>(att, sa_wo + wofs, D_, D_, x, x, nullptr, 0);
        // ---- cross attention ----
        k_rms<true><<<M_, 256, 0, stream>>>(x, ca_g + l * D_, (void*)hbf);
        k_gemm<2><<<g1024, 128, 0, stream>>>(hbf,  ca_wq + wofs, D_, D_, nullptr, nullptr, qb,   0);
        k_gemm<2><<<g1024, 128, 0, stream>>>(memb, ca_wk + wofs, D_, D_, nullptr, nullptr, kbuf, 0);
        k_gemm<3><<<g1024, 128, 0, stream>>>(memb, ca_wv + wofs, D_, D_, nullptr, nullptr, vbuf, 0);
        k_attn<<<gattn, 32, 0, stream>>>(qb, kbuf, vbuf, att, 0);
        k_gemm<0><<<g1024, 128, 0, stream>>>(att, ca_wo + wofs, D_, D_, x, x, nullptr, 0);
        // ---- MLP ----
        k_rms<true><<<M_, 256, 0, stream>>>(x, mlp_g + l * D_, (void*)hbf);
        k_gemm<1><<<g4096, 128, 0, stream>>>(hbf, w1 + (size_t)l * D_ * FFN_, FFN_, D_,
                                             nullptr, nullptr, ffn, 1);
        k_gemm<0><<<g1024, 128, 0, stream>>>(ffn, w2 + (size_t)l * FFN_ * D_, D_, FFN_,
                                             x, x, nullptr, 0);
    }
    k_rms<false><<<M_, 256, 0, stream>>>(x, fin_g, (void*)out);
}
